// LanguageModel_13091060318262
// MI455X (gfx1250) — compile-verified, matrix operand order checked by reading, not run on hardware
//
#include <hip/hip_runtime.h>
#include <hip/hip_bf16.h>

// ---------------------------------------------------------------------------
// 2-layer LSTM LM:  embed -> LSTM1 -> LSTM2 -> W_out proj -> log_softmax
// VOC=32000 EMB=1024 HID=1024 B=32 S=128
// Strategy: bf16 WMMA for all GEMMs, f32 state. Output projection streams
// W_out panels through LDS with the Tensor Data Mover (tensor_load_to_lds,
// TENSORcnt), double-buffered, 2 WMMA tiles per wave.
// ---------------------------------------------------------------------------

#define VOC 32000
#define EMB 1024
#define HID 1024
#define BB  32
#define SS  128

typedef __attribute__((ext_vector_type(16))) __bf16 v16bf;
typedef __attribute__((ext_vector_type(8)))  __bf16 v8bf;
typedef __attribute__((ext_vector_type(8)))  float  v8f;
typedef __attribute__((ext_vector_type(4)))  unsigned int v4u;
typedef __attribute__((ext_vector_type(8)))  int v8i;
typedef __attribute__((ext_vector_type(4)))  int v4i;

// Load a 16x32 bf16 tile (row-major, leading dim `ld`) into WMMA A/B fragment
// layout: lanes 0-15 hold rows 0-15 with K = {0..7, 16..23};
//         lanes 16-31 hold rows 0-15 with K = {8..15, 24..31}.
__device__ __forceinline__ v16bf load_frag16x32(const __bf16* base, int ld) {
  int lane = threadIdx.x & 31;
  int row  = lane & 15;
  int kh   = (lane >> 4) << 3;           // 0 or 8
  const __bf16* p = base + (size_t)row * ld + kh;
  v8bf lo = *(const v8bf*)(p);           // K = kh .. kh+7      (16B aligned)
  v8bf hi = *(const v8bf*)(p + 16);      // K = kh+16 .. kh+23  (16B aligned)
  v16bf a;
#pragma unroll
  for (int i = 0; i < 8; ++i) { a[i] = lo[i]; a[i + 8] = hi[i]; }
  return a;
}

// ------------------------- Tensor Data Mover (TDM) -------------------------
// Issue a 2D bf16 tile DMA (tile_d1 rows of tile_d0 elements, row stride
// stride0 elements) from global to LDS. D# bitfields per CDNA5 ISA §8.3/8.4.
// Wave-scalar op: tracked by TENSORcnt, EXEC ignored.
__device__ __forceinline__ void tdm_load_2d_bf16(unsigned lds_addr,
                                                 unsigned long long gaddr,
                                                 unsigned tile_d0,
                                                 unsigned tile_d1,
                                                 unsigned stride0) {
  v4u g0;
  g0[0] = 1u;                                        // count=1, user mode
  g0[1] = lds_addr;                                  // LDS byte address
  g0[2] = (unsigned)(gaddr & 0xffffffffu);           // global_addr[31:0]
  g0[3] = (unsigned)((gaddr >> 32) & 0x1ffffffu)     // global_addr[56:32]
        | (2u << 30);                                // type = 2 ("image")
  v8i g1;
  g1[0] = (int)(1u << 16);                           // wg_mask=0, data_size=2B
  g1[1] = (int)((tile_d0 & 0xffffu) << 16);          // tensor_dim0[15:0]
  g1[2] = (int)(((tile_d0 >> 16) & 0xffffu)          // tensor_dim0[31:16]
        | ((tile_d1 & 0xffffu) << 16));              // tensor_dim1[15:0]
  g1[3] = (int)(((tile_d1 >> 16) & 0xffffu)          // tensor_dim1[31:16]
        | ((tile_d0 & 0xffffu) << 16));              // tile_dim0
  g1[4] = (int)(tile_d1 & 0xffffu);                  // tile_dim1, tile_dim2=0
  g1[5] = (int)stride0;                              // tensor_dim0_stride lo
  g1[6] = 0;                                         // stride hi, dim1_stride
  g1[7] = 0;
  v4i gz = {0, 0, 0, 0};
#if defined(__clang_major__) && (__clang_major__ >= 23)
  v8i gz8 = {0, 0, 0, 0, 0, 0, 0, 0};
  __builtin_amdgcn_tensor_load_to_lds(g0, g1, gz, gz, gz8, 0);
#else
  __builtin_amdgcn_tensor_load_to_lds(g0, g1, gz, gz, 0);
#endif
}

// Generic (flat) pointer to a __shared__ object: low 32 bits are the LDS
// byte offset (flat LDS aperture truncates addr[31:0]).
__device__ __forceinline__ unsigned lds_byte_addr(const void* p) {
  return (unsigned)(unsigned long long)p;
}

// ---------------------------- preprocessing --------------------------------

__global__ void embed_gather_bf16(const long long* __restrict__ idx,
                                  const float* __restrict__ emb,
                                  __bf16* __restrict__ xbuf) {
  int bs = blockIdx.x;                   // b*S + s
  int b = bs / SS, s = bs % SS;
  long long row = idx[bs];
  const float* src = emb + (size_t)row * EMB;
  __bf16* dst = xbuf + ((size_t)s * BB + b) * EMB;   // x laid out [S][B][E]
  for (int e = threadIdx.x; e < EMB; e += blockDim.x)
    dst[e] = (__bf16)src[e];
}

// Wcat[n][0..1023] = W_ih[n][:], Wcat[n][1024..2047] = W_hh[n][:]   (bf16)
__global__ void cat_convert_bf16(__bf16* __restrict__ dst,
                                 const float* __restrict__ wa,
                                 const float* __restrict__ wb,
                                 size_t rows) {
  size_t total = rows * 2048;
  for (size_t i = (size_t)blockIdx.x * blockDim.x + threadIdx.x; i < total;
       i += (size_t)gridDim.x * blockDim.x) {
    size_t n = i >> 11;
    size_t k = i & 2047;
    float v = (k < 1024) ? wa[n * 1024 + k] : wb[n * 1024 + (k - 1024)];
    dst[i] = (__bf16)v;
  }
}

__global__ void convert_bf16(__bf16* __restrict__ dst,
                             const float* __restrict__ src, size_t n) {
  for (size_t i = (size_t)blockIdx.x * blockDim.x + threadIdx.x; i < n;
       i += (size_t)gridDim.x * blockDim.x)
    dst[i] = (__bf16)src[i];
}

__global__ void bias_sum(float* __restrict__ dst, const float* __restrict__ a,
                         const float* __restrict__ b, int n) {
  int i = blockIdx.x * blockDim.x + threadIdx.x;
  if (i < n) dst[i] = a[i] + b[i];
}

__global__ void zero_f32(float* p, size_t n) {
  for (size_t i = (size_t)blockIdx.x * blockDim.x + threadIdx.x; i < n;
       i += (size_t)gridDim.x * blockDim.x) p[i] = 0.0f;
}
__global__ void zero_bf16(__bf16* p, size_t n) {
  for (size_t i = (size_t)blockIdx.x * blockDim.x + threadIdx.x; i < n;
       i += (size_t)gridDim.x * blockDim.x) p[i] = (__bf16)0.0f;
}

// ------------------------------ LSTM step ----------------------------------
// gates[32,4096] = [x|h][32,2048] @ Wcat^T + bias ; then fused i,f,g,o math.
// Grid: 64 blocks (one per 16 columns of H). Block: 256 threads = 8 waves.
// Wave w -> (mt = w&1 batch-tile, gate = w>>1). Each wave: one 16x16 WMMA tile.
__global__ void __launch_bounds__(256)
lstm_step(const __bf16* __restrict__ x,      // [32][1024] bf16
          const __bf16* __restrict__ hprev,  // [32][1024] bf16
          __bf16* __restrict__ hnew,         // [32][1024] bf16
          float* __restrict__ c,             // [32][1024] f32
          const __bf16* __restrict__ Wcat,   // [4096][2048] bf16
          const float* __restrict__ bias,    // [4096] = b_ih+b_hh
          __bf16* __restrict__ hseq_out,     // [B*S][1024] or null
          int t) {
  __shared__ float lds[8][16][16];

  int wave = threadIdx.x >> 5;
  int lane = threadIdx.x & 31;
  int mt   = wave & 1;                 // batch tile (rows 0-15 / 16-31)
  int gate = wave >> 1;                // i, f, g, o
  int jt   = blockIdx.x;               // H column tile
  int nrow = gate * HID + jt * 16;     // row of Wcat = gate column of output

  const __bf16* wbase = Wcat + (size_t)nrow * 2048;
  const __bf16* xrow  = x + (size_t)mt * 16 * 1024;
  const __bf16* hrow  = hprev + (size_t)mt * 16 * 1024;

  v8f acc = {};
#pragma unroll 4
  for (int ks = 0; ks < 32; ++ks) {    // x half: K = 0..1023
    v16bf afrag = load_frag16x32(xrow + ks * 32, 1024);
    v16bf bfrag = load_frag16x32(wbase + ks * 32, 2048);
    acc = __builtin_amdgcn_wmma_f32_16x16x32_bf16(
        false, afrag, false, bfrag, (short)0, acc, false, false);
  }
#pragma unroll 4
  for (int ks = 0; ks < 32; ++ks) {    // h half: K = 1024..2047
    v16bf afrag = load_frag16x32(hrow + ks * 32, 1024);
    v16bf bfrag = load_frag16x32(wbase + 1024 + ks * 32, 2048);
    acc = __builtin_amdgcn_wmma_f32_16x16x32_bf16(
        false, afrag, false, bfrag, (short)0, acc, false, false);
  }

  // Scatter C tile into LDS: VGPR r -> row r (lanes 0-15) / row 8+r (16-31).
  {
    int col = lane & 15;
    int rb  = (lane >> 4) << 3;
#pragma unroll
    for (int r = 0; r < 8; ++r) lds[wave][rb + r][col] = acc[r];
  }
  __syncthreads();

  // Fused elementwise: 32 batch x 16 cols = 512 outputs.
  for (int e = threadIdx.x; e < 512; e += blockDim.x) {
    int b   = e >> 4;
    int cj  = e & 15;
    int mti = b >> 4;
    int row = b & 15;
    int j   = jt * 16 + cj;
    float gi = lds[(0 << 1) | mti][row][cj] + bias[0 * HID + j];
    float gf = lds[(1 << 1) | mti][row][cj] + bias[1 * HID + j];
    float gg = lds[(2 << 1) | mti][row][cj] + bias[2 * HID + j];
    float go = lds[(3 << 1) | mti][row][cj] + bias[3 * HID + j];
    float si = 1.0f / (1.0f + __expf(-gi));
    float sf = 1.0f / (1.0f + __expf(-gf));
    float so = 1.0f / (1.0f + __expf(-go));
    float tg = tanhf(gg);
    size_t off = (size_t)b * HID + j;
    float cn = sf * c[off] + si * tg;
    c[off] = cn;
    float hn = so * tanhf(cn);
    hnew[off] = (__bf16)hn;
    if (hseq_out)
      hseq_out[((size_t)b * SS + t) * HID + j] = (__bf16)hn;
  }
}

// --------------------------- output projection -----------------------------
// logits[4096, 32000] = hseq[4096,1024] @ Wout^T + b_out
// Block tile: M=32 x N=128.  8 waves: wave -> (mrow = w&1, ng = w>>1),
// each wave owns a 16x32 output strip (2 WMMA accumulators).
// W_out panel (128 rows x 32 K, bf16 = 8KB) is DMA'd into LDS by the TDM,
// double-buffered; wave 0 issues tensor_load_to_lds and gates TENSORcnt.
__global__ void __launch_bounds__(256)
logits_gemm(const __bf16* __restrict__ hseq,
            const __bf16* __restrict__ wout,  // [32000][1024] bf16
            const float* __restrict__ bout,
            float* __restrict__ out) {
  __shared__ __bf16 Bsh[2][128 * 32];          // 2 x 8KB double buffer

  int wave = threadIdx.x >> 5;
  int lane = threadIdx.x & 31;
  int mt   = blockIdx.y;                       // 0..127  (32-row M block)
  int nb   = blockIdx.x;                       // 0..249  (128-col N block)
  int mrow = wave & 1;                         // 16-row half
  int ng   = wave >> 1;                        // 0..3 -> 32-col strip

  const __bf16* abase = hseq + ((size_t)mt * 32 + (size_t)mrow * 16) * 1024;
  unsigned long long gbase =
      (unsigned long long)(const void*)(wout + (size_t)nb * 128 * 1024);

  v8f acc0 = {}, acc1 = {};

  if (wave == 0)                               // prologue: panel ks=0
    tdm_load_2d_bf16(lds_byte_addr(&Bsh[0][0]), gbase, 32u, 128u, 1024u);

  for (int ks = 0; ks < 32; ++ks) {            // K = 1024 in 32-wide panels
    if (wave == 0) {
      if (ks + 1 < 32) {                       // issue next panel, then wait
        tdm_load_2d_bf16(lds_byte_addr(&Bsh[(ks + 1) & 1][0]),
                         gbase + (unsigned long long)(ks + 1) * 64u,
                         32u, 128u, 1024u);
        __builtin_amdgcn_s_wait_tensorcnt(1);  // current panel complete
      } else {
        __builtin_amdgcn_s_wait_tensorcnt(0);
      }
    }
    __syncthreads();                           // panel visible to all waves

    const __bf16* bpan = &Bsh[ks & 1][0];
    v16bf afrag = load_frag16x32(abase + ks * 32, 1024);
    v16bf b0 = load_frag16x32(bpan + (size_t)(ng * 32) * 32, 32);
    v16bf b1 = load_frag16x32(bpan + (size_t)(ng * 32 + 16) * 32, 32);
    acc0 = __builtin_amdgcn_wmma_f32_16x16x32_bf16(
        false, afrag, false, b0, (short)0, acc0, false, false);
    acc1 = __builtin_amdgcn_wmma_f32_16x16x32_bf16(
        false, afrag, false, b1, (short)0, acc1, false, false);

    __syncthreads();                           // done reading before overwrite
  }

  int rb   = (lane >> 4) << 3;
  int row0 = mt * 32 + mrow * 16 + rb;
  int col0 = nb * 128 + ng * 32 + (lane & 15);
  int col1 = col0 + 16;
  float bv0 = bout[col0], bv1 = bout[col1];
#pragma unroll
  for (int r = 0; r < 8; ++r) {
    size_t rowoff = (size_t)(row0 + r) * VOC;
    out[rowoff + col0] = acc0[r] + bv0;
    out[rowoff + col1] = acc1[r] + bv1;
  }
}

// ------------------------------ log_softmax --------------------------------
__global__ void logsoftmax_rows(float* __restrict__ out) {
  __shared__ float red[256];
  float* p = out + (size_t)blockIdx.x * VOC;

  float m = -3.0e38f;
  for (int v = threadIdx.x; v < VOC; v += blockDim.x) m = fmaxf(m, p[v]);
  red[threadIdx.x] = m;
  __syncthreads();
  for (int s = 128; s > 0; s >>= 1) {
    if (threadIdx.x < s)
      red[threadIdx.x] = fmaxf(red[threadIdx.x], red[threadIdx.x + s]);
    __syncthreads();
  }
  m = red[0];
  __syncthreads();

  float sum = 0.0f;
  for (int v = threadIdx.x; v < VOC; v += blockDim.x) sum += __expf(p[v] - m);
  red[threadIdx.x] = sum;
  __syncthreads();
  for (int s = 128; s > 0; s >>= 1) {
    if (threadIdx.x < s) red[threadIdx.x] += red[threadIdx.x + s];
    __syncthreads();
  }
  float lse = m + __logf(red[0]);

  for (int v = threadIdx.x; v < VOC; v += blockDim.x) p[v] -= lse;
}

// ------------------------------- launcher ----------------------------------
extern "C" void kernel_launch(void* const* d_in, const int* in_sizes, int n_in,
                              void* d_out, int out_size, void* d_ws,
                              size_t ws_size, hipStream_t stream) {
  const long long* inputs = (const long long*)d_in[0];
  const float* emb   = (const float*)d_in[1];
  const float* W_ih1 = (const float*)d_in[2];
  const float* W_hh1 = (const float*)d_in[3];
  const float* b_ih1 = (const float*)d_in[4];
  const float* b_hh1 = (const float*)d_in[5];
  const float* W_ih2 = (const float*)d_in[6];
  const float* W_hh2 = (const float*)d_in[7];
  const float* b_ih2 = (const float*)d_in[8];
  const float* b_hh2 = (const float*)d_in[9];
  const float* W_out = (const float*)d_in[10];
  const float* b_out = (const float*)d_in[11];
  float* out = (float*)d_out;

  // workspace carve-up (256B aligned)
  char* ws = (char*)d_ws;
  size_t off = 0;
  auto carve = [&](size_t bytes) -> void* {
    void* p = ws + off;
    off = (off + bytes + 255) & ~(size_t)255;
    return p;
  };
  __bf16* xbuf  = (__bf16*)carve((size_t)SS * BB * EMB * 2);       //  8 MB
  __bf16* Wcat1 = (__bf16*)carve((size_t)4 * HID * 2048 * 2);      // 16 MB
  __bf16* Wcat2 = (__bf16*)carve((size_t)4 * HID * 2048 * 2);      // 16 MB
  __bf16* WoutB = (__bf16*)carve((size_t)VOC * HID * 2);           // 64 MB
  float*  bsum1 = (float*)carve((size_t)4 * HID * 4);
  float*  bsum2 = (float*)carve((size_t)4 * HID * 4);
  __bf16* h1a   = (__bf16*)carve((size_t)BB * HID * 2);
  __bf16* h1b   = (__bf16*)carve((size_t)BB * HID * 2);
  __bf16* h2a   = (__bf16*)carve((size_t)BB * HID * 2);
  __bf16* h2b   = (__bf16*)carve((size_t)BB * HID * 2);
  float*  c1    = (float*)carve((size_t)BB * HID * 4);
  float*  c2    = (float*)carve((size_t)BB * HID * 4);
  __bf16* hseq  = (__bf16*)carve((size_t)BB * SS * HID * 2);       //  8 MB
  __bf16* h1buf[2] = {h1a, h1b};
  __bf16* h2buf[2] = {h2a, h2b};

  // --- preprocessing: bf16 weights, fused biases, zero state, embed gather
  cat_convert_bf16<<<2048, 256, 0, stream>>>(Wcat1, W_ih1, W_hh1, 4 * HID);
  cat_convert_bf16<<<2048, 256, 0, stream>>>(Wcat2, W_ih2, W_hh2, 4 * HID);
  convert_bf16<<<4096, 256, 0, stream>>>(WoutB, W_out, (size_t)VOC * HID);
  bias_sum<<<16, 256, 0, stream>>>(bsum1, b_ih1, b_hh1, 4 * HID);
  bias_sum<<<16, 256, 0, stream>>>(bsum2, b_ih2, b_hh2, 4 * HID);
  zero_f32<<<64, 256, 0, stream>>>(c1, (size_t)BB * HID);
  zero_f32<<<64, 256, 0, stream>>>(c2, (size_t)BB * HID);
  zero_bf16<<<64, 256, 0, stream>>>(h1a, (size_t)BB * HID);
  zero_bf16<<<64, 256, 0, stream>>>(h2a, (size_t)BB * HID);
  embed_gather_bf16<<<BB * SS, 256, 0, stream>>>(inputs, emb, xbuf);

  // --- recurrence: 2 fused WMMA+activation launches per time step
  for (int t = 0; t < SS; ++t) {
    const __bf16* x_t = xbuf + (size_t)t * BB * EMB;
    lstm_step<<<HID / 16, 256, 0, stream>>>(
        x_t, h1buf[t & 1], h1buf[(t + 1) & 1], c1, Wcat1, bsum1,
        (__bf16*)nullptr, t);
    lstm_step<<<HID / 16, 256, 0, stream>>>(
        h1buf[(t + 1) & 1], h2buf[t & 1], h2buf[(t + 1) & 1], c2, Wcat2, bsum2,
        hseq, t);
  }

  // --- output projection (TDM-staged B panels) + log_softmax
  logits_gemm<<<dim3(VOC / 128, (BB * SS) / 32), 256, 0, stream>>>(
      hseq, WoutB, b_out, out);
  logsoftmax_rows<<<BB * SS, 256, 0, stream>>>(out);
}
